// HomogeneousDeepGraphEmbedding1_18322330484882
// MI455X (gfx1250) — compile-verified, matrix operand order checked by reading, not run on hardware
//
#include <hip/hip_runtime.h>
#include <stdint.h>

#ifndef ATHENA_ASYNC
#define ATHENA_ASYNC 1
#endif

#define N_NODES   16384
#define N_EDGES   262144
#define F_IN      300
#define F_IN_PAD  320
#define BSH       256
#define H2        128
#define H4        64
#define H8        32
#define NGRAPH    32
#define NODES_PER_G (N_NODES / NGRAPH)
#define OUTF      8
#define EPSV      1e-5f
#define ALPHAV    0.5f
#define BETAV     0.04879016416943205f   /* log(0.1/2 + 1) */

typedef __attribute__((ext_vector_type(16))) __bf16 v16bf;
typedef __attribute__((ext_vector_type(8)))  float  v8f;

union Frag16 { uint4 q[2]; uint32_t u[8]; v16bf v; };

__device__ __forceinline__ uint16_t f2bf(float f) {
  uint32_t u = __float_as_uint(f);
  u += 0x7fffu + ((u >> 16) & 1u);           // round-to-nearest-even
  return (uint16_t)(u >> 16);
}
__device__ __forceinline__ unsigned ordf(float f) {   // order-preserving uint encoding
  unsigned u = __float_as_uint(f);
  return (u & 0x80000000u) ? ~u : (u | 0x80000000u);
}
__device__ __forceinline__ float iordf(unsigned u) {
  unsigned v = (u & 0x80000000u) ? (u ^ 0x80000000u) : ~u;
  return __uint_as_float(v);
}

/* ------------------------------------------------------------------ */
/* zero fills                                                          */
__global__ __launch_bounds__(256) void k_zero_f(float* p, size_t n) {
  size_t t = (size_t)blockIdx.x * 256 + threadIdx.x;
  if (t < n) p[t] = 0.0f;
}
__global__ __launch_bounds__(256) void k_zero_u(unsigned* p, size_t n) {
  size_t t = (size_t)blockIdx.x * 256 + threadIdx.x;
  if (t < n) p[t] = 0u;
}

/* ------------------------------------------------------------------ */
/* f32 -> bf16 conversions                                             */
__global__ __launch_bounds__(256) void k_cvt_flat_pad(const float* __restrict__ X,
                                                      uint16_t* __restrict__ Y,
                                                      size_t nvalid, size_t ntot) {
  size_t t = (size_t)blockIdx.x * 256 + threadIdx.x;
  if (t < ntot) Y[t] = (t < nvalid) ? f2bf(X[t]) : (uint16_t)0;
}
__global__ __launch_bounds__(256) void k_cvt_rowpad(const float* __restrict__ X,
                                                    uint16_t* __restrict__ Y,
                                                    int K, int Kp, size_t ntot) {
  size_t t = (size_t)blockIdx.x * 256 + threadIdx.x;
  if (t >= ntot) return;
  int r = (int)(t / Kp), c = (int)(t % Kp);
  Y[t] = (c < K) ? f2bf(X[(size_t)r * K + c]) : (uint16_t)0;
}
/* W[K x M] f32 -> Wt[M x Kp] bf16 (transposed, K padded with zeros)    */
__global__ __launch_bounds__(256) void k_cvt_w_t(const float* __restrict__ W,
                                                 uint16_t* __restrict__ Y,
                                                 int K, int Kp, int M, size_t ntot) {
  size_t t = (size_t)blockIdx.x * 256 + threadIdx.x;
  if (t >= ntot) return;
  int col = (int)(t / Kp), k = (int)(t % Kp);
  Y[t] = (k < K) ? f2bf(W[(size_t)k * M + col]) : (uint16_t)0;
}

/* ------------------------------------------------------------------ */
/* WMMA bf16 GEMM:  C[rows x M] = A[rows x Kp] * B[Kp x M]             */
/* A row-major bf16; B supplied TRANSPOSED (Bt[M x Kp] bf16).          */
/* block = 256 threads (8 waves); wave -> 16x32 tile; B panel in LDS   */
/* staged column-major via async b128 copies; frags load as b128.      */
__global__ __launch_bounds__(256)
void k_gemm_bf16_wmma(const uint16_t* __restrict__ A,
                      const uint16_t* __restrict__ Bt,
                      float* __restrict__ C,
                      int rows, int Kp, int M) {
  extern __shared__ __align__(16) uint16_t sB[];   // 32 cols, stride SP elems
  const int SP = Kp + 8;                           // keeps 16B alignment per col
  const int c0 = blockIdx.y * 32;

  // Stage B panel (32 columns of Kp bf16) into LDS, 16B chunks
  const int cchunks = Kp >> 3;                     // 16B chunks per column
  const int nchunk = 32 * cchunks;
  for (int idx = threadIdx.x; idx < nchunk; idx += 256) {
    int col = idx / cchunks;
    int kk  = (idx - col * cchunks) << 3;
    const uint16_t* gp = Bt + (size_t)(c0 + col) * Kp + kk;
#if ATHENA_ASYNC
    uint32_t lds_addr = (uint32_t)(uintptr_t)(sB + (size_t)col * SP + kk);
    asm volatile("global_load_async_to_lds_b128 %0, %1, off"
                 :: "v"(lds_addr), "v"(gp) : "memory");
#else
    *(uint4*)(sB + (size_t)col * SP + kk) = *(const uint4*)gp;
#endif
  }
#if ATHENA_ASYNC
  asm volatile("s_wait_asynccnt 0" ::: "memory");
#endif
  __syncthreads();

  const int wave = threadIdx.x >> 5;
  const int lane = threadIdx.x & 31;
  const int r0 = (blockIdx.x * 8 + wave) * 16;
  if (r0 >= rows) return;                          // no barriers after this point
  const int m = lane & 15, half = lane >> 4;

  v8f acc0 = {}; v8f acc1 = {};
  const uint16_t* arow  = A + (size_t)(r0 + m) * Kp + 8 * half;
  const uint16_t* bcol0 = sB + (size_t)m * SP + 8 * half;
  const uint16_t* bcol1 = sB + (size_t)(16 + m) * SP + 8 * half;

  for (int k = 0; k < Kp; k += 32) {
    Frag16 a, b0, b1;
    a.q[0]  = *(const uint4*)(arow + k);           // K = 8h .. 8h+7
    a.q[1]  = *(const uint4*)(arow + k + 16);      // K = 16+8h .. 16+8h+7
    b0.q[0] = *(const uint4*)(bcol0 + k);
    b0.q[1] = *(const uint4*)(bcol0 + k + 16);
    b1.q[0] = *(const uint4*)(bcol1 + k);
    b1.q[1] = *(const uint4*)(bcol1 + k + 16);
    acc0 = __builtin_amdgcn_wmma_f32_16x16x32_bf16(false, a.v, false, b0.v,
                                                   (short)0, acc0, false, false);
    acc1 = __builtin_amdgcn_wmma_f32_16x16x32_bf16(false, a.v, false, b1.v,
                                                   (short)0, acc1, false, false);
  }
#pragma unroll
  for (int r = 0; r < 8; ++r) {
    const int row = r0 + r + 8 * half;
    C[(size_t)row * M + c0 + m]      = acc0[r];
    C[(size_t)row * M + c0 + 16 + m] = acc1[r];
  }
}

/* ------------------------------------------------------------------ */
/* GCN normalization precompute                                        */
__global__ __launch_bounds__(256) void k_deg(const int* __restrict__ dst,
                                             const float* __restrict__ w,
                                             float* degw, float* deg1) {
  size_t t = (size_t)blockIdx.x * 256 + threadIdx.x;
  if (t < N_EDGES) {
    int d = dst[t];
    unsafeAtomicAdd(&degw[d], w[t]);
    unsafeAtomicAdd(&deg1[d], 1.0f);
  } else if (t < (size_t)N_EDGES + N_NODES) {
    int i = (int)(t - N_EDGES);
    unsafeAtomicAdd(&degw[i], 1.0f);
    unsafeAtomicAdd(&deg1[i], 1.0f);
  }
}
__device__ __forceinline__ float dinv(float d) {
  return d > 0.0f ? rsqrtf(fmaxf(d, 1e-12f)) : 0.0f;
}
__global__ __launch_bounds__(256) void k_norm(const int* __restrict__ src,
                                              const int* __restrict__ dst,
                                              const float* __restrict__ w,
                                              const float* degw, const float* deg1,
                                              float* normw, float* norm1,
                                              float* selfw, float* self1) {
  size_t t = (size_t)blockIdx.x * 256 + threadIdx.x;
  if (t < N_EDGES) {
    int s = src[t], d = dst[t];
    normw[t] = dinv(degw[s]) * w[t] * dinv(degw[d]);
    norm1[t] = dinv(deg1[s]) * dinv(deg1[d]);
  } else if (t < (size_t)N_EDGES + N_NODES) {
    int i = (int)(t - N_EDGES);
    float a = dinv(degw[i]); selfw[i] = a * a;
    float b = dinv(deg1[i]); self1[i] = b * b;
  }
}

/* ------------------------------------------------------------------ */
/* graph aggregation: O = selfn*H  then  O[dst] += H[src]*norm         */
__global__ __launch_bounds__(256) void k_agg_init(const float* __restrict__ H,
                                                  const float* __restrict__ selfn,
                                                  float* __restrict__ O,
                                                  int cshift, size_t tot) {
  size_t t = (size_t)blockIdx.x * 256 + threadIdx.x;
  if (t >= tot) return;
  size_t i = t >> cshift;
  O[t] = H[t] * selfn[i];
}
__global__ __launch_bounds__(256) void k_agg_edges(const float* __restrict__ H,
                                                   const int* __restrict__ src,
                                                   const int* __restrict__ dst,
                                                   const float* __restrict__ norm,
                                                   float* __restrict__ O,
                                                   int cshift, size_t tot) {
  size_t t = (size_t)blockIdx.x * 256 + threadIdx.x;
  if (t >= tot) return;
  size_t e = t >> cshift;
  int c = (int)(t & (((size_t)1 << cshift) - 1));
  int s = src[e], d = dst[e];
  float v = H[((size_t)s << cshift) + c] * norm[e];
  unsafeAtomicAdd(&O[((size_t)d << cshift) + c], v);
}

/* ------------------------------------------------------------------ */
/* BatchNorm (training stats) + ReLU, fused bf16 emit                  */
__global__ __launch_bounds__(256) void k_bn_stats(const float* __restrict__ X,
                                                  int C, float* mu, float* rstd) {
  int c = blockIdx.x;
  float s = 0.f, s2 = 0.f;
  for (int i = threadIdx.x; i < N_NODES; i += 256) {
    float v = X[(size_t)i * C + c];
    s += v; s2 += v * v;
  }
  __shared__ float sa[256], sb[256];
  sa[threadIdx.x] = s; sb[threadIdx.x] = s2;
  __syncthreads();
  for (int o = 128; o > 0; o >>= 1) {
    if (threadIdx.x < o) { sa[threadIdx.x] += sa[threadIdx.x + o]; sb[threadIdx.x] += sb[threadIdx.x + o]; }
    __syncthreads();
  }
  if (threadIdx.x == 0) {
    float m = sa[0] / (float)N_NODES;
    float var = sb[0] / (float)N_NODES - m * m;
    mu[c] = m; rstd[c] = rsqrtf(var + EPSV);
  }
}
__global__ __launch_bounds__(256) void k_bn_relu(const float* __restrict__ X,
                                                 const float* __restrict__ mu,
                                                 const float* __restrict__ rstd,
                                                 const float* __restrict__ g,
                                                 const float* __restrict__ be,
                                                 float* __restrict__ Yf,
                                                 uint16_t* __restrict__ Ybf,
                                                 int cshift, size_t tot) {
  size_t t = (size_t)blockIdx.x * 256 + threadIdx.x;
  if (t >= tot) return;
  int c = (int)(t & (((size_t)1 << cshift) - 1));
  float y = (X[t] - mu[c]) * rstd[c] * g[c] + be[c];
  y = fmaxf(y, 0.0f);
  Yf[t] = y;
  if (Ybf) Ybf[t] = f2bf(y);
}

/* ------------------------------------------------------------------ */
/* GCN2 blends                                                         */
__global__ __launch_bounds__(256) void k_gcn2_h(const float* agg, const float* x0,
                                                float* h, size_t tot) {
  size_t t = (size_t)blockIdx.x * 256 + threadIdx.x;
  if (t < tot) h[t] = (1.0f - ALPHAV) * agg[t] + ALPHAV * x0[t];
}
__global__ __launch_bounds__(256) void k_gcn2_out(const float* h, const float* hW,
                                                  float* o, size_t tot) {
  size_t t = (size_t)blockIdx.x * 256 + threadIdx.x;
  if (t < tot) o[t] = (1.0f - BETAV) * h[t] + BETAV * hW[t];
}

/* ------------------------------------------------------------------ */
/* GATv2 (heads = 2)                                                   */
__global__ __launch_bounds__(256) void k_gat_alpha(const float* __restrict__ xl,
                                                   const float* __restrict__ xr,
                                                   const float* __restrict__ att,
                                                   const int* __restrict__ src,
                                                   const int* __restrict__ dst,
                                                   float* __restrict__ alpha,
                                                   unsigned* __restrict__ amax,
                                                   int CH) {
  size_t t = (size_t)blockIdx.x * 256 + threadIdx.x;
  if (t >= (size_t)N_EDGES + N_NODES) return;
  int s, d;
  if (t < N_EDGES) { s = src[t]; d = dst[t]; } else { s = d = (int)(t - N_EDGES); }
  const int HC = 2 * CH;
  for (int hd = 0; hd < 2; ++hd) {
    float a = 0.f;
    for (int c = 0; c < CH; ++c) {
      float v = xl[(size_t)s * HC + hd * CH + c] + xr[(size_t)d * HC + hd * CH + c];
      v = (v > 0.f) ? v : 0.2f * v;
      a += v * att[hd * CH + c];
    }
    alpha[t * 2 + hd] = a;
    atomicMax(&amax[(size_t)d * 2 + hd], ordf(a));
  }
}
__global__ __launch_bounds__(256) void k_gat_exp(const int* __restrict__ dst,
                                                 const float* __restrict__ alpha,
                                                 const unsigned* __restrict__ amax,
                                                 float* __restrict__ ex,
                                                 float* __restrict__ den) {
  size_t t = (size_t)blockIdx.x * 256 + threadIdx.x;
  if (t >= (size_t)N_EDGES + N_NODES) return;
  int d = (t < N_EDGES) ? dst[t] : (int)(t - N_EDGES);
  for (int hd = 0; hd < 2; ++hd) {
    float e = __expf(alpha[t * 2 + hd] - iordf(amax[(size_t)d * 2 + hd]));
    ex[t * 2 + hd] = e;
    unsafeAtomicAdd(&den[(size_t)d * 2 + hd], e);
  }
}
__global__ __launch_bounds__(256) void k_gat_agg(const int* __restrict__ src,
                                                 const int* __restrict__ dst,
                                                 const float* __restrict__ xl,
                                                 const float* __restrict__ ex,
                                                 const float* __restrict__ den,
                                                 float* __restrict__ O,
                                                 int CH, int hcshift, size_t tot) {
  size_t t = (size_t)blockIdx.x * 256 + threadIdx.x;
  if (t >= tot) return;
  size_t e = t >> hcshift;
  int c = (int)(t & (((size_t)1 << hcshift) - 1));
  int hd = c / CH;
  int s, d;
  if (e < N_EDGES) { s = src[e]; d = dst[e]; } else { s = d = (int)(e - N_EDGES); }
  float a = ex[e * 2 + hd] / (den[(size_t)d * 2 + hd] + 1e-16f);
  unsafeAtomicAdd(&O[((size_t)d << hcshift) + c], xl[((size_t)s << hcshift) + c] * a);
}

/* ------------------------------------------------------------------ */
/* MemPooling                                                          */
__global__ __launch_bounds__(256) void k_x2(const float* __restrict__ X, float* x2n) {
  size_t t = (size_t)blockIdx.x * 256 + threadIdx.x;
  if (t >= N_NODES) return;
  float s = 0.f;
  for (int c = 0; c < BSH; ++c) { float v = X[t * BSH + c]; s += v * v; }
  x2n[t] = s;
}
__global__ void k_keys(const float* __restrict__ key, float* k2) {
  int t = threadIdx.x;
  if (t >= 8) return;
  float s = 0.f;
  for (int c = 0; c < BSH; ++c) { float v = key[t * BSH + c]; s += v * v; }
  k2[t] = s;
}
__global__ __launch_bounds__(256) void k_xk(const float* __restrict__ X,
                                            const float* __restrict__ key,
                                            float* __restrict__ xk) {
  size_t t = (size_t)blockIdx.x * 256 + threadIdx.x;
  if (t >= (size_t)N_NODES * 8) return;
  size_t n = t >> 3; int hk = (int)(t & 7);
  float s = 0.f;
  for (int c = 0; c < BSH; ++c) s += X[n * BSH + c] * key[hk * BSH + c];
  xk[t] = s;
}
__global__ __launch_bounds__(256) void k_assign(const float* __restrict__ x2n,
                                                const float* __restrict__ k2,
                                                const float* __restrict__ xk,
                                                const float* __restrict__ cw,
                                                const float* __restrict__ cb,
                                                float* __restrict__ S2) {
  size_t t = (size_t)blockIdx.x * 256 + threadIdx.x;
  if (t >= N_NODES) return;
  float Sa[4][2];
  for (int hk = 0; hk < 8; ++hk) {
    float dd = fmaxf(x2n[t] + k2[hk] - 2.0f * xk[t * 8 + hk], 0.0f);
    Sa[hk >> 1][hk & 1] = 1.0f / (1.0f + dd);
  }
  float c0 = cb[0], c1 = cb[0];
  for (int h = 0; h < 4; ++h) {
    float inv = 1.0f / (Sa[h][0] + Sa[h][1]);
    c0 += cw[h] * Sa[h][0] * inv;
    c1 += cw[h] * Sa[h][1] * inv;
  }
  float m = fmaxf(c0, c1);
  float e0 = __expf(c0 - m), e1 = __expf(c1 - m);
  float inv = 1.0f / (e0 + e1);
  S2[t * 2 + 0] = e0 * inv;
  S2[t * 2 + 1] = e1 * inv;
}
__global__ __launch_bounds__(256) void k_pool(const float* __restrict__ X,
                                              const float* __restrict__ S2,
                                              float* __restrict__ pooled) {
  int b = blockIdx.x >> 1, k = blockIdx.x & 1, c = threadIdx.x;
  float s = 0.f;
  for (int j = 0; j < NODES_PER_G; ++j) {
    size_t n = (size_t)b * NODES_PER_G + j;
    s += S2[n * 2 + k] * X[n * BSH + c];
  }
  pooled[(size_t)blockIdx.x * BSH + c] = s;
}
__global__ void k_final(const float* __restrict__ flat64,
                        const float* __restrict__ lb,
                        const float* __restrict__ W,
                        const float* __restrict__ bias,
                        float* __restrict__ out) {
  int t = threadIdx.x;                        // 256 = B * OUTF
  if (t >= NGRAPH * OUTF) return;
  int b = t >> 3, o = t & 7;
  float s = bias[o];
  for (int c = 0; c < BSH; ++c) {
    float v = flat64[(size_t)(b * 2 + (c >> 7)) * H2 + (c & 127)] + lb[c & 127];
    s += v * W[c * OUTF + o];
  }
  out[t] = s;
}

/* ================================================================== */
extern "C" void kernel_launch(void* const* d_in, const int* in_sizes, int n_in,
                              void* d_out, int out_size, void* d_ws, size_t ws_size,
                              hipStream_t stream) {
  (void)in_sizes; (void)n_in; (void)out_size; (void)ws_size;

  const float* x   = (const float*)d_in[0];
  const float* ew  = (const float*)d_in[1];
  const int*   ei  = (const int*)d_in[98];
  const int*   src = ei;
  const int*   dst = ei + N_EDGES;

  auto P = [&](int i) { return (const float*)d_in[i]; };

  /* workspace carve-out */
  char* wp = (char*)d_ws;
  auto alloc = [&](size_t bytes) -> void* {
    void* r = (void*)wp;
    wp += (bytes + 255) & ~(size_t)255;
    return r;
  };
  float*    degw  = (float*)alloc((size_t)N_NODES * 4);
  float*    deg1  = (float*)alloc((size_t)N_NODES * 4);
  float*    normw = (float*)alloc((size_t)N_EDGES * 4);
  float*    norm1 = (float*)alloc((size_t)N_EDGES * 4);
  float*    selfw = (float*)alloc((size_t)N_NODES * 4);
  float*    self1 = (float*)alloc((size_t)N_NODES * 4);
  uint16_t* bfA   = (uint16_t*)alloc((size_t)N_NODES * F_IN_PAD * 2);
  uint16_t* bfB   = (uint16_t*)alloc((size_t)N_NODES * F_IN_PAD * 2);
  uint16_t* bfW   = (uint16_t*)alloc((size_t)F_IN_PAD * BSH * 2);
  float*    tmp1  = (float*)alloc((size_t)N_NODES * BSH * 4);
  float*    tmp2  = (float*)alloc((size_t)N_NODES * BSH * 4);
  float*    hbuf1 = (float*)alloc((size_t)N_NODES * BSH * 4);
  float*    hbuf2 = (float*)alloc((size_t)N_NODES * BSH * 4);
  float*    x1    = (float*)alloc((size_t)N_NODES * BSH * 4);
  float*    x2s   = (float*)alloc((size_t)N_NODES * H2 * 4);
  float*    x3s   = (float*)alloc((size_t)N_NODES * H4 * 4);
  float*    x4s   = (float*)alloc((size_t)N_NODES * H8 * 4);
  float*    mu    = (float*)alloc(BSH * 4);
  float*    rstd  = (float*)alloc(BSH * 4);
  float*    xl    = (float*)alloc((size_t)N_NODES * H2 * 4);
  float*    xr    = (float*)alloc((size_t)N_NODES * H2 * 4);
  float*    galp  = (float*)alloc((size_t)(N_EDGES + N_NODES) * 2 * 4);
  float*    gex   = (float*)alloc((size_t)(N_EDGES + N_NODES) * 2 * 4);
  unsigned* amax  = (unsigned*)alloc((size_t)N_NODES * 2 * 4);
  float*    den   = (float*)alloc((size_t)N_NODES * 2 * 4);
  float*    x2n   = (float*)alloc((size_t)N_NODES * 4);
  float*    k2    = (float*)alloc(8 * 4);
  float*    xk    = (float*)alloc((size_t)N_NODES * 8 * 4);
  float*    S2    = (float*)alloc((size_t)N_NODES * 2 * 4);
  float*    pooled= (float*)alloc((size_t)NGRAPH * 2 * BSH * 4);
  uint16_t* pbf   = (uint16_t*)alloc((size_t)NGRAPH * 2 * BSH * 2);
  float*    flat64= (float*)alloc((size_t)NGRAPH * 2 * H2 * 4);

  auto blocks = [](size_t tot) { return dim3((unsigned)((tot + 255) / 256)); };
  auto ilog2  = [](int v) { int s = 0; while ((1 << s) < v) ++s; return s; };

  auto zeroF = [&](float* p, size_t n) { k_zero_f<<<blocks(n), 256, 0, stream>>>(p, n); };
  auto zeroU = [&](unsigned* p, size_t n) { k_zero_u<<<blocks(n), 256, 0, stream>>>(p, n); };

  auto gemm = [&](const uint16_t* A, const uint16_t* Bt, float* C, int rows, int Kp, int M) {
    dim3 g((unsigned)((rows + 127) / 128), (unsigned)(M / 32));
    size_t shb = (size_t)(Kp + 8) * 32 * 2;
    k_gemm_bf16_wmma<<<g, 256, shb, stream>>>(A, Bt, C, rows, Kp, M);
  };
  auto cvtW = [&](const float* Wf, int K, int Kp, int M) {   // -> bfW transposed [M x Kp]
    size_t tot = (size_t)Kp * M;
    k_cvt_w_t<<<blocks(tot), 256, 0, stream>>>(Wf, bfW, K, Kp, M, tot);
  };
  auto cvtAct = [&](const float* Xf, uint16_t* Y, size_t tot) {
    k_cvt_flat_pad<<<blocks(tot), 256, 0, stream>>>(Xf, Y, tot, tot);
  };
  auto agg = [&](const float* H, const float* norm, const float* selfn, float* O, int C) {
    int cs = ilog2(C);
    size_t totn = (size_t)N_NODES * C;
    size_t tote = (size_t)N_EDGES * C;
    k_agg_init<<<blocks(totn), 256, 0, stream>>>(H, selfn, O, cs, totn);
    k_agg_edges<<<blocks(tote), 256, 0, stream>>>(H, src, dst, norm, O, cs, tote);
  };
  auto bn = [&](const float* Xf, int C, const float* g, const float* be,
                float* outF, uint16_t* outBf) {
    k_bn_stats<<<C, 256, 0, stream>>>(Xf, C, mu, rstd);
    size_t tot = (size_t)N_NODES * C;
    k_bn_relu<<<blocks(tot), 256, 0, stream>>>(Xf, mu, rstd, g, be, outF, outBf, ilog2(C), tot);
  };
  auto gcn = [&](const uint16_t* act, int Kp, const float* Wf, int Kin, int C,
                 const float* g, const float* be, const float* norm, const float* selfn,
                 float* outF, uint16_t* outBf) {
    cvtW(Wf, Kin, Kp, C);
    gemm(act, bfW, tmp1, N_NODES, Kp, C);
    agg(tmp1, norm, selfn, tmp2, C);
    bn(tmp2, C, g, be, outF, outBf);
  };
  auto gcn2l = [&](const float* xin, const float* x0f, const float* Wf, int C,
                   const float* g, const float* be, float* outF, uint16_t* outBf,
                   uint16_t* scratchBf) {
    size_t tot = (size_t)N_NODES * C;
    agg(xin, normw, selfw, tmp1, C);
    k_gcn2_h<<<blocks(tot), 256, 0, stream>>>(tmp1, x0f, hbuf2, tot);
    cvtAct(hbuf2, scratchBf, tot);
    cvtW(Wf, C, C, C);
    gemm(scratchBf, bfW, tmp1, N_NODES, C, C);
    k_gcn2_out<<<blocks(tot), 256, 0, stream>>>(hbuf2, tmp1, tmp2, tot);
    bn(tmp2, C, g, be, outF, outBf);
  };
  auto gat = [&](const uint16_t* actBf, int Kin, int CH,
                 const float* Wl, const float* Wr, const float* attv,
                 const float* g, const float* be, float* outF, uint16_t* outBf) {
    const int HC = 2 * CH;
    cvtW(Wl, Kin, Kin, HC);
    gemm(actBf, bfW, xl, N_NODES, Kin, HC);
    cvtW(Wr, Kin, Kin, HC);
    gemm(actBf, bfW, xr, N_NODES, Kin, HC);
    size_t totE = (size_t)N_EDGES + N_NODES;
    zeroU(amax, (size_t)N_NODES * 2);
    zeroF(den, (size_t)N_NODES * 2);
    k_gat_alpha<<<blocks(totE), 256, 0, stream>>>(xl, xr, attv, src, dst, galp, amax, CH);
    k_gat_exp<<<blocks(totE), 256, 0, stream>>>(dst, galp, amax, gex, den);
    size_t tot = totE * HC;
    zeroF(tmp1, (size_t)N_NODES * HC);
    k_gat_agg<<<blocks(tot), 256, 0, stream>>>(src, dst, xl, gex, den, tmp1, CH, ilog2(HC), tot);
    bn(tmp1, HC, g, be, outF, outBf);
  };

  /* ---- gcn_norm (weighted + unweighted) ---- */
  zeroF(degw, N_NODES); zeroF(deg1, N_NODES);
  size_t totEN = (size_t)N_EDGES + N_NODES;
  k_deg<<<blocks(totEN), 256, 0, stream>>>(dst, ew, degw, deg1);
  k_norm<<<blocks(totEN), 256, 0, stream>>>(src, dst, ew, degw, deg1,
                                            normw, norm1, selfw, self1);

  /* ---- encoder: 12 x (GCNConv -> BN -> ReLU), skip taps at 3/6/9/12 ---- */
  {
    size_t totx = (size_t)N_NODES * F_IN_PAD;
    k_cvt_rowpad<<<blocks(totx), 256, 0, stream>>>(x, bfA, F_IN, F_IN_PAD, totx);
  }
  const int encIn[12]  = {300, 256, 256, 256, 128, 128, 128, 64, 64, 64, 32, 32};
  const int encOut[12] = {256, 256, 256, 128, 128, 128, 64, 64, 64, 32, 32, 32};
  uint16_t* cur = bfA;
  uint16_t* nxt = bfB;
  for (int L = 0; L < 12; ++L) {
    int Kin = encIn[L], Kp = ((Kin + 31) / 32) * 32, C = encOut[L];
    float* fdst = (L == 2) ? x1 : (L == 5) ? x2s : (L == 8) ? x3s : (L == 11) ? x4s : hbuf1;
    gcn(cur, Kp, P(2 + 4 * L), Kin, C, P(4 + 4 * L), P(5 + 4 * L),
        normw, selfw, fdst, nxt);
    uint16_t* t = cur; cur = nxt; nxt = t;
  }
  uint16_t* bfX = cur;   // bf16 of x4
  uint16_t* bfY = nxt;

  /* ---- attention block ---- */
  gat(bfX, H8, H8, P(50), P(51), P(52), P(54), P(55), hbuf1, bfY);        // gat1 -> h4 [N,H4]
  gcn2l(hbuf1, x3s, P(56), H4, P(57), P(58), hbuf1, bfY, bfX);            // gcn2  -> h3
  gcn(bfY, H4, P(59), H4, H4, P(61), P(62), normw, selfw, hbuf1, bfX);    // gcn   -> h3
  gat(bfX, H4, H4, P(63), P(64), P(65), P(67), P(68), hbuf1, bfY);        // gat2 -> h3 [N,H2]

  /* ---- decoder block ---- */
  gcn2l(hbuf1, x2s, P(69), H2, P(70), P(71), hbuf1, bfY, bfX);            // gcn2a
  gcn(bfY, H2, P(72), H2, H2, P(74), P(75), normw, selfw, hbuf1, bfX);    // gcn1
  gcn(bfX, H2, P(76), H2, BSH, P(78), P(79), norm1, self1, hbuf1, bfY);   // gcn2 (unweighted)
  gcn2l(hbuf1, x1, P(80), BSH, P(81), P(82), hbuf1, bfX, bfY);            // gcn2b
  gcn(bfX, BSH, P(83), BSH, BSH, P(85), P(86), normw, selfw, hbuf1, bfY); // gcn3
  gcn(bfY, BSH, P(87), BSH, BSH, P(89), P(90), normw, selfw, hbuf1, bfX); // gcn4 -> h1

  /* ---- MemPooling + output ---- */
  k_x2<<<blocks(N_NODES), 256, 0, stream>>>(hbuf1, x2n);
  k_keys<<<1, 8, 0, stream>>>(P(91), k2);
  k_xk<<<blocks((size_t)N_NODES * 8), 256, 0, stream>>>(hbuf1, P(91), xk);
  k_assign<<<blocks(N_NODES), 256, 0, stream>>>(x2n, k2, xk, P(92), P(93), S2);
  k_pool<<<NGRAPH * 2, BSH, 0, stream>>>(hbuf1, S2, pooled);
  cvtAct(pooled, pbf, (size_t)NGRAPH * 2 * BSH);
  cvtW(P(94), BSH, BSH, H2);
  gemm(pbf, bfW, flat64, NGRAPH * 2, BSH, H2);
  k_final<<<1, 256, 0, stream>>>(flat64, P(95), P(96), P(97), (float*)d_out);
}
